// ColBERT_23837068492859
// MI455X (gfx1250) — matrix operand order repaired and checked by reading, not compiled, not run on hardware
//
#include <hip/hip_runtime.h>
#include <hip/hip_bf16.h>
#include <stdint.h>

// Problem constants (match reference)
constexpr int B_   = 64;
constexpr int LQ_  = 128;
constexpr int LD_  = 512;
constexpr int H_   = 768;
constexpr int DIM_ = 128;

typedef __bf16 v16bf __attribute__((ext_vector_type(16)));
typedef float  v8f   __attribute__((ext_vector_type(8)));

__device__ __forceinline__ unsigned short f32_to_bf16_rne(float f) {
    union { float f; unsigned int u; } x;
    x.f = f;
    unsigned int u = x.u;
    u += 0x7FFFu + ((u >> 16) & 1u);   // round-to-nearest-even
    return (unsigned short)(u >> 16);
}

// ---------------------------------------------------------------------------
// Kernel 0: W [H, DIM] fp32  ->  Wt [DIM, H] bf16  (B-operand becomes rows)
// ---------------------------------------------------------------------------
__global__ void wt_convert_kernel(const float* __restrict__ W,
                                  unsigned short* __restrict__ Wt) {
    int idx = blockIdx.x * blockDim.x + threadIdx.x;
    if (idx >= DIM_ * H_) return;
    int d = idx / H_;
    int h = idx - d * H_;
    Wt[idx] = f32_to_bf16_rne(W[(size_t)h * DIM_ + d]);
}

// ---------------------------------------------------------------------------
// Kernel 1: projection + (mask) + L2-normalize, fp32 hidden -> bf16 out.
// Block = 16 rows x 128 cols (8 waves, one 16-col slab each).
// K loop: 12 iterations of 2x(16x16x32) WMMA; A chunks staged fp32->bf16 in
// double-buffered LDS (one barrier per iteration, staging overlaps WMMA).
// Epilogue: C tile -> padded LDS, cooperative row-norms, coalesced 16B stores.
// ---------------------------------------------------------------------------
template <bool USE_MASK>
__global__ __launch_bounds__(256)
void proj_norm_kernel(const float* __restrict__ hidden,
                      const unsigned short* __restrict__ Wt,
                      const int* __restrict__ mask,
                      unsigned short* __restrict__ outp) {
    constexpr int LSTR   = 40;          // bf16 elems; 80B row stride (conflict-free)
    constexpr int NPAIR  = H_ / 64;     // 12
    __shared__ __align__(16) unsigned short atile[2][2][16 * LSTR];
    __shared__ __align__(16) float cbuf[16][132];   // padded: rows p,p+8 hit diff banks
    __shared__ float psum[16][16];
    __shared__ float rowscale[16];

    const int wave = threadIdx.x >> 5;
    const int lane = threadIdx.x & 31;
    const int half = lane >> 4;
    const int l15  = lane & 15;
    const int R    = blockIdx.x * 16;   // global row tile base
    const int C    = wave * 16;         // output column base for this wave

    const unsigned short* brow = Wt + (size_t)(C + l15) * H_;
    const int kb   = half * 8;   // B: lanes 0-15 -> K {0..7,16..23}, 16-31 -> {8..15,24..31}
    const int asel = half * 8;   // A: lanes 0-15 -> K {0..7,16..23}, 16-31 -> {8..15,24..31}

    // staging assignment: 16 rows x 64 K-vals (2 chunks) per iteration,
    // 4 fp32 -> 4 bf16 per thread
    const int se  = threadIdx.x * 4;    // 0..1020
    const int sr  = se >> 6;            // row 0..15
    const int sk  = se & 63;            // K offset in pair, multiple of 4
    const int sch = sk >> 5;            // chunk 0/1
    const int skc = sk & 31;
    const float* srow = hidden + (size_t)(R + sr) * H_;
    unsigned int* const sdst[2] = {
        (unsigned int*)(atile[0][sch] + sr * LSTR + skc),
        (unsigned int*)(atile[1][sch] + sr * LSTR + skc)
    };

    auto stage = [&](int pair) {
        float4 f4 = *(const float4*)(srow + pair * 64 + sk);
        unsigned int* dst = sdst[pair & 1];
        dst[0] = ((unsigned int)f32_to_bf16_rne(f4.y) << 16) | f32_to_bf16_rne(f4.x);
        dst[1] = ((unsigned int)f32_to_bf16_rne(f4.w) << 16) | f32_to_bf16_rne(f4.z);
    };

    stage(0);

    v8f c = {};
    for (int pair = 0; pair < NPAIR; ++pair) {
        __syncthreads();
        if (pair + 1 < NPAIR) stage(pair + 1);      // overlaps with WMMA below
#pragma unroll
        for (int ch = 0; ch < 2; ++ch) {
            const unsigned short* arow = atile[pair & 1][ch] + l15 * LSTR;
            union { uint4 u[2]; v16bf v; } af;
            af.u[0] = *(const uint4*)(arow + asel);
            af.u[1] = *(const uint4*)(arow + 16 + asel);

            const int k0 = pair * 64 + ch * 32;
            union { uint4 u[2]; v16bf v; } bf;
            bf.u[0] = *(const uint4*)(brow + k0 + kb);
            bf.u[1] = *(const uint4*)(brow + k0 + kb + 16);

            c = __builtin_amdgcn_wmma_f32_16x16x32_bf16(
                    false, af.v, false, bf.v, (short)0, c, false, false);
        }
    }
    __syncthreads();

    // C tile -> LDS (VGPR p, lanes 0-15 -> row p, lanes 16-31 -> row p+8)
#pragma unroll
    for (int p = 0; p < 8; ++p)
        cbuf[p + half * 8][C + l15] = c[p];
    __syncthreads();

    // cooperative row sum-of-squares: 16 threads per row, 8 elems each
    {
        const int row = threadIdx.x >> 4;
        const int seg = threadIdx.x & 15;
        float s = 0.0f;
#pragma unroll
        for (int j = 0; j < 8; ++j) {
            float v = cbuf[row][seg * 8 + j];
            s += v * v;
        }
        psum[row][seg] = s;
    }
    __syncthreads();
    if (threadIdx.x < 16) {
        float s = 0.0f;
#pragma unroll
        for (int j = 0; j < 16; ++j) s += psum[threadIdx.x][j];
        float scale = 1.0f / fmaxf(sqrtf(s), 1e-12f);
        if constexpr (USE_MASK) scale *= (float)mask[R + threadIdx.x];
        rowscale[threadIdx.x] = scale;
    }
    __syncthreads();

    // coalesced store: each thread writes 8 consecutive bf16 (one 16B uint4)
    {
        const int row = threadIdx.x >> 4;
        const int cs  = (threadIdx.x & 15) * 8;
        const float s = rowscale[row];
        union { unsigned short h[8]; uint4 u; } o;
#pragma unroll
        for (int j = 0; j < 8; ++j)
            o.h[j] = f32_to_bf16_rne(cbuf[row][cs + j] * s);
        *(uint4*)(outp + (size_t)(R + row) * DIM_ + cs) = o.u;
    }
}

// ---------------------------------------------------------------------------
// Kernel 2: maxsim. Block = (batch b, 16 q-rows). 8 waves x 4 doc tiles each
// (fully unrolled -> 16 WMMAs per wave, A frags preloaded in registers).
// Epilogue: partial-max tiles -> padded LDS, cooperative max + sum, atomicAdd.
// ---------------------------------------------------------------------------
__global__ __launch_bounds__(256)
void maxsim_kernel(const unsigned short* __restrict__ Qn, // [B*LQ, DIM] bf16
                   const unsigned short* __restrict__ Dn, // [B*LD, DIM] bf16
                   float* __restrict__ score) {           // [B]
    __shared__ __align__(16) float mbuf[16][132];
    __shared__ float pmax[16][16];
    __shared__ float rowmax[16];

    const int wave = threadIdx.x >> 5;
    const int lane = threadIdx.x & 31;
    const int half = lane >> 4;
    const int l15  = lane & 15;
    const int b    = blockIdx.x >> 3;   // LQ/16 == 8 q-tiles per batch
    const int qt   = blockIdx.x & 7;

    const int asel = half * 8;
    const int kb   = half * 8;

    // Preload A fragments (16 q rows x full K=128)
    const unsigned short* qrow = Qn + (size_t)(b * LQ_ + qt * 16 + l15) * DIM_;
    v16bf a[4];
#pragma unroll
    for (int kk = 0; kk < 4; ++kk) {
        union { uint4 u[2]; v16bf v; } f;
        f.u[0] = *(const uint4*)(qrow + kk * 32 + asel);
        f.u[1] = *(const uint4*)(qrow + kk * 32 + 16 + asel);
        a[kk] = f.v;
    }

    v8f runmax;
#pragma unroll
    for (int p = 0; p < 8; ++p) runmax[p] = -3.0e38f;

#pragma unroll
    for (int i = 0; i < 4; ++i) {                    // uniform trip count
        const int t = wave + i * 8;                  // doc tile 0..31
        const unsigned short* drow = Dn + (size_t)(b * LD_ + t * 16 + l15) * DIM_;
        v8f c = {};
#pragma unroll
        for (int kk = 0; kk < 4; ++kk) {
            union { uint4 u[2]; v16bf v; } f;
            f.u[0] = *(const uint4*)(drow + kk * 32 + kb);
            f.u[1] = *(const uint4*)(drow + kk * 32 + kb + 16);
            c = __builtin_amdgcn_wmma_f32_16x16x32_bf16(
                    false, a[kk], false, f.v, (short)0, c, false, false);
        }
#pragma unroll
        for (int p = 0; p < 8; ++p) runmax[p] = fmaxf(runmax[p], c[p]);
    }

    // per-wave partial-max tile -> LDS slots [row][wave*16 + col]
#pragma unroll
    for (int p = 0; p < 8; ++p)
        mbuf[p + half * 8][wave * 16 + l15] = runmax[p];
    __syncthreads();

    // cooperative max over 128 slots per q-row
    {
        const int row = threadIdx.x >> 4;
        const int seg = threadIdx.x & 15;
        float m = -3.0e38f;
#pragma unroll
        for (int j = 0; j < 8; ++j) m = fmaxf(m, mbuf[row][seg * 8 + j]);
        pmax[row][seg] = m;
    }
    __syncthreads();
    if (threadIdx.x < 16) {
        float m = pmax[threadIdx.x][0];
#pragma unroll
        for (int j = 1; j < 16; ++j) m = fmaxf(m, pmax[threadIdx.x][j]);
        rowmax[threadIdx.x] = m;
    }
    __syncthreads();
    if (threadIdx.x == 0) {
        float s = 0.0f;
#pragma unroll
        for (int r = 0; r < 16; ++r) s += rowmax[r];
        atomicAdd(&score[b], s);
    }
}

// ---------------------------------------------------------------------------
extern "C" void kernel_launch(void* const* d_in, const int* in_sizes, int n_in,
                              void* d_out, int out_size, void* d_ws, size_t ws_size,
                              hipStream_t stream) {
    const float* q_hidden = (const float*)d_in[0]; // [B, LQ, H]
    const float* d_hidden = (const float*)d_in[1]; // [B, LD, H]
    const float* W        = (const float*)d_in[2]; // [H, DIM]
    const int*   doc_mask = (const int*)d_in[3];   // [B, LD]
    float* out = (float*)d_out;                    // [B]

    // workspace layout (bf16 halves): Wt | Qn | Dn   (~10.4 MB total)
    unsigned short* Wt = (unsigned short*)d_ws;                 // DIM*H
    unsigned short* Qn = Wt + (size_t)DIM_ * H_;                // B*LQ*DIM
    unsigned short* Dn = Qn + (size_t)B_ * LQ_ * DIM_;          // B*LD*DIM

    wt_convert_kernel<<<(DIM_ * H_ + 255) / 256, 256, 0, stream>>>(W, Wt);

    proj_norm_kernel<false><<<B_ * LQ_ / 16, 256, 0, stream>>>(
        q_hidden, Wt, nullptr, Qn);
    proj_norm_kernel<true><<<B_ * LD_ / 16, 256, 0, stream>>>(
        d_hidden, Wt, doc_mask, Dn);

    hipMemsetAsync(d_out, 0, (size_t)B_ * sizeof(float), stream);

    maxsim_kernel<<<B_ * (LQ_ / 16), 256, 0, stream>>>(Qn, Dn, out);
}
